// RelGraphConvHetero_52690658787907
// MI455X (gfx1250) — compile-verified, matrix operand order checked by reading, not run on hardware
//
#include <hip/hip_runtime.h>
#include <hip/hip_bf16.h>

// ---------------------------------------------------------------------------
// RGCN hetero forward, edge-centric fused WMMA version for gfx1250 (MI455X).
// Y[dst, :] += W[rel] * X[src, :]  per edge, F_in = F_out = 128, 8 relations.
// ---------------------------------------------------------------------------

typedef __attribute__((ext_vector_type(16))) __bf16 bf16x16;
typedef __attribute__((ext_vector_type(8)))  float  floatx8;

union FragBF {
    bf16x16  v;
    unsigned u[8];
};

#define F_DIM   128
#define N_RELS  8

__device__ __forceinline__ unsigned short f32_to_bf16_rne(float f) {
    unsigned u = __float_as_uint(f);
    unsigned rounding = 0x7FFFu + ((u >> 16) & 1u);
    return (unsigned short)((u + rounding) >> 16);
}

__device__ __forceinline__ unsigned pack_bf16x2(float lo, float hi) {
    return (unsigned)f32_to_bf16_rne(lo) | ((unsigned)f32_to_bf16_rne(hi) << 16);
}

// K index inside a 32-wide K-block for 16-bit WMMA A/B fragments (ISA 7.12.2):
// lanes 0-15: dwords 0-3 -> K=0..7, dwords 4-7 -> K=16..23
// lanes 16-31: dwords 0-3 -> K=8..15, dwords 4-7 -> K=24..31
__device__ __forceinline__ int frag_k(int hi, int d) {
    return (d < 4) ? (8 * hi + 2 * d) : (16 + 8 * hi + 2 * (d - 4));
}

// ---------------------------------------------------------------------------
// K0: zero output Y and the workspace control block (counters/offsets).
// ---------------------------------------------------------------------------
__global__ void k_zero(float* __restrict__ Y, int n, int* __restrict__ ctrl) {
    int i = blockIdx.x * 256 + threadIdx.x;
    if (i < n) Y[i] = 0.0f;
    if (i < 64) ctrl[i] = 0;
}

// ---------------------------------------------------------------------------
// K1: histogram edges per relation.
// ---------------------------------------------------------------------------
__global__ void k_hist(const int* __restrict__ edge_type, int E,
                       int* __restrict__ counts) {
    int e = blockIdx.x * 256 + threadIdx.x;
    if (e < E) atomicAdd(&counts[edge_type[e]], 1);
}

// ---------------------------------------------------------------------------
// K2: tiny exclusive scan over 8 relation counts (1 thread).
// ---------------------------------------------------------------------------
__global__ void k_scan(int* __restrict__ ws) {
    int* counts  = ws;        // [8]
    int* cursors = ws + 8;    // [8]
    int* offsets = ws + 16;   // [9]
    int* tileOff = ws + 32;   // [9]
    offsets[0] = 0;
    tileOff[0] = 0;
    for (int r = 0; r < N_RELS; ++r) {
        offsets[r + 1] = offsets[r] + counts[r];
        tileOff[r + 1] = tileOff[r] + (counts[r] + 15) / 16;
        cursors[r]     = offsets[r];
    }
}

// ---------------------------------------------------------------------------
// K3: scatter edge ids into relation bins.
// ---------------------------------------------------------------------------
__global__ void k_scatter(const int* __restrict__ edge_type, int E,
                          int* __restrict__ cursors, int* __restrict__ binned) {
    int e = blockIdx.x * 256 + threadIdx.x;
    if (e < E) {
        int r   = edge_type[e];
        int pos = atomicAdd(&cursors[r], 1);
        binned[pos] = e;
    }
}

// ---------------------------------------------------------------------------
// K4: repack W[r, o, f] (f32) into bf16 B-fragment dword layout:
//   Bpacked[((r*8 + ob)*4 + kb)*32*8 + lane*8 + d] holds the packed pair
//   (W[r][o][k], W[r][o][k+1]) with o = ob*16 + (lane&15),
//   k = kb*32 + frag_k(lane>>4, d).   Total 64K dwords = 256 KB, L2-resident.
// ---------------------------------------------------------------------------
__global__ void k_packW(const float* __restrict__ W, unsigned* __restrict__ Bp) {
    int idx  = blockIdx.x * 256 + threadIdx.x;   // < 8*8*4*32*8 = 65536
    int d    = idx & 7;
    int lane = (idx >> 3) & 31;
    int kb   = (idx >> 8) & 3;
    int ob   = (idx >> 10) & 7;
    int r    = idx >> 13;
    int hi   = lane >> 4;
    int n    = lane & 15;
    int k    = kb * 32 + frag_k(hi, d);
    int o    = ob * 16 + n;
    const float* w = W + ((size_t)(r * F_DIM + o)) * F_DIM + k;
    Bp[idx] = pack_bf16x2(w[0], w[1]);
}

// ---------------------------------------------------------------------------
// K5: main fused kernel. One block = one 16-edge same-relation tile.
//   - stage 16 gathered X rows into LDS in A-fragment dword layout (f32->bf16)
//   - 4 waves, each owning 2 output 16-col blocks; K-loop of 4x32
//   - 8x v_wmma_f32_16x16x32_bf16 per wave, then masked f32 atomic scatter
// ---------------------------------------------------------------------------
__global__ __launch_bounds__(128) void k_rgcn_tile(
        const float* __restrict__ X,
        const int*   __restrict__ edge_src,
        const int*   __restrict__ edge_dst,
        const int*   __restrict__ ws_ctrl,     // counts/cursors/offsets/tileOff
        const int*   __restrict__ binned,
        const unsigned* __restrict__ Bpacked,
        float* __restrict__ Y) {
    __shared__ unsigned sA[4][32][8];          // A fragments, 4 KB
    __shared__ int s_src[16];
    __shared__ int s_dst[16];
    __shared__ int s_meta[3];                  // rel, valid, base

    const int tid  = threadIdx.x;
    const int lane = tid & 31;
    const int wv   = tid >> 5;                 // wave id 0..3
    const int hi   = lane >> 4;
    const int n    = lane & 15;

    const int* offsets = ws_ctrl + 16;
    const int* tileOff = ws_ctrl + 32;

    if (tid == 0) {
        int b     = blockIdx.x;
        int total = tileOff[N_RELS];
        if (b >= total) {
            s_meta[0] = -1;
        } else {
            int r = 0;
            while (r < N_RELS - 1 && b >= tileOff[r + 1]) r++;
            int lt   = b - tileOff[r];
            int base = offsets[r] + lt * 16;
            int rem  = offsets[r + 1] - base;
            s_meta[0] = r;
            s_meta[1] = (rem < 16) ? rem : 16;
            s_meta[2] = base;
        }
    }
    __syncthreads();
    const int rel = s_meta[0];
    if (rel < 0) return;
    const int valid = s_meta[1];
    const int base  = s_meta[2];

    if (tid < 16) {
        int m  = tid;
        int mc = (m < valid) ? m : (valid - 1);  // clamp padding rows to a valid edge
        int e  = binned[base + mc];
        s_src[m] = edge_src[e];
        s_dst[m] = (m < valid) ? edge_dst[e] : -1;
    }
    __syncthreads();

    // Stage gathered X rows (f32 -> packed bf16 pairs) into A-fragment layout.
    {
        int kb = tid >> 5;                      // 0..3
        int m  = n;                             // row of A this lane services
        const float* xrow = X + (size_t)s_src[m] * F_DIM;
        #pragma unroll
        for (int d = 0; d < 8; ++d) {
            int k = kb * 32 + frag_k(hi, d);
            float2 xv = *(const float2*)(xrow + k);   // k even -> 8B aligned
            sA[kb][lane][d] = pack_bf16x2(xv.x, xv.y);
        }
    }
    __syncthreads();

    // Each wave computes output column blocks {wv, wv+4}.
    const int ob0 = wv;
    const int ob1 = wv + 4;
    floatx8 c0 = {};
    floatx8 c1 = {};

    #pragma unroll
    for (int kb = 0; kb < 4; ++kb) {
        FragBF a, b0, b1;
        #pragma unroll
        for (int d = 0; d < 8; ++d) a.u[d] = sA[kb][lane][d];

        const unsigned* bp0 =
            Bpacked + ((((size_t)rel * 8 + ob0) * 4 + kb) * 32 + lane) * 8;
        const unsigned* bp1 =
            Bpacked + ((((size_t)rel * 8 + ob1) * 4 + kb) * 32 + lane) * 8;
        #pragma unroll
        for (int d = 0; d < 8; ++d) { b0.u[d] = bp0[d]; b1.u[d] = bp1[d]; }

        c0 = __builtin_amdgcn_wmma_f32_16x16x32_bf16(
                 false, a.v, false, b0.v, (short)0, c0, false, false);
        c1 = __builtin_amdgcn_wmma_f32_16x16x32_bf16(
                 false, a.v, false, b1.v, (short)0, c1, false, false);
    }

    // Scatter-add: C/D layout (ISA 7.12.2): element (m, n) at VGPR v with
    // m = v + 8*hi, n = lane&15. Padding rows have dst == -1 and are skipped.
    #pragma unroll
    for (int v = 0; v < 8; ++v) {
        int m   = v + 8 * hi;
        int dst = s_dst[m];
        if (dst >= 0) {
            float* yrow = Y + (size_t)dst * F_DIM;
            unsafeAtomicAdd(yrow + ob0 * 16 + n, c0[v]);
            unsafeAtomicAdd(yrow + ob1 * 16 + n, c1[v]);
        }
    }
}

// ---------------------------------------------------------------------------
// Host-side launcher.
// ---------------------------------------------------------------------------
extern "C" void kernel_launch(void* const* d_in, const int* in_sizes, int n_in,
                              void* d_out, int out_size, void* d_ws, size_t ws_size,
                              hipStream_t stream) {
    const float* X         = (const float*)d_in[0];
    const float* W         = (const float*)d_in[1];
    const int*   edge_src  = (const int*)d_in[2];
    const int*   edge_dst  = (const int*)d_in[3];
    const int*   edge_type = (const int*)d_in[4];
    float*       Y         = (float*)d_out;

    const int E = in_sizes[2];

    // Workspace layout (ints):
    //   [0..7]   counts      [8..15]  cursors
    //   [16..24] offsets(9)  [32..40] tileOffsets(9)
    //   [64..64+E)           binned edge ids
    //   after (16B aligned)  Bpacked: 65536 dwords (256 KB) of bf16 B-fragments
    int* ws_i = (int*)d_ws;
    int* counts  = ws_i;
    int* cursors = ws_i + 8;
    int* binned  = ws_i + 64;
    size_t bpOff = 64 + (size_t)((E + 3) & ~3);
    unsigned* Bpacked = (unsigned*)(ws_i + bpOff);

    // K0: zero Y + control block
    {
        int n = out_size;
        int blocks = (n + 255) / 256;
        k_zero<<<blocks, 256, 0, stream>>>(Y, n, ws_i);
    }
    // K1: histogram
    k_hist<<<(E + 255) / 256, 256, 0, stream>>>(edge_type, E, counts);
    // K2: scan
    k_scan<<<1, 1, 0, stream>>>(ws_i);
    // K3: scatter into bins
    k_scatter<<<(E + 255) / 256, 256, 0, stream>>>(edge_type, E, cursors, binned);
    // K4: repack W to bf16 B-fragments (65536 dwords)
    k_packW<<<65536 / 256, 256, 0, stream>>>(W, Bpacked);
    // K5: main fused WMMA kernel; upper bound on tiles = E/16 + one partial per rel
    {
        int maxTiles = (E + 15) / 16 + N_RELS;
        k_rgcn_tile<<<maxTiles, 128, 0, stream>>>(
            X, edge_src, edge_dst, ws_i, binned, Bpacked, Y);
    }
}
